// CustomConv_84859963834659
// MI455X (gfx1250) — matrix operand order repaired and checked by reading, not compiled, not run on hardware
//
#include <hip/hip_runtime.h>
#include <math.h>

typedef __attribute__((ext_vector_type(16))) __bf16 v16bf;
typedef __attribute__((ext_vector_type(8)))  float  v8f;

#define N_NODES 100000
#define N_EDGES 1600000
#define D 128

// Round-to-nearest-even f32 -> bf16 without relying on __bf16 cast codegen.
__device__ __forceinline__ __bf16 f2bf(float f) {
  unsigned int u = __float_as_uint(f);
  unsigned int r = (u + 0x7FFFu + ((u >> 16) & 1u)) >> 16;
  return __builtin_bit_cast(__bf16, (unsigned short)r);
}

// ---------------------------------------------------------------------------
// h = x @ W  via v_wmma_f32_16x16x32_bf16.  One wave = 16 rows x 128 cols.
// W (128x128) staged transposed into LDS as bf16 so B fragments are
// contiguous 32B chunks -> ds_load_b128 pairs.
// ---------------------------------------------------------------------------
__global__ __launch_bounds__(256) void gcn_gemm_kernel(
    const float* __restrict__ x, const float* __restrict__ W,
    float* __restrict__ h) {
  __shared__ __align__(32) __bf16 Wt[D * D];  // Wt[n*D + k] = W[k*D + n]
  const int tid = threadIdx.x;
  for (int i = tid; i < D * D; i += 256) {
    int k = i >> 7, n = i & 127;
    Wt[n * D + k] = f2bf(W[i]);
  }
  __syncthreads();

  const int wave = tid >> 5;
  const int lane = tid & 31;
  const int rowTile = blockIdx.x * 8 + wave;
  if (rowTile >= (N_NODES / 16)) return;  // uniform per wave; EXEC stays all-1s
  const int rowbase = rowTile * 16;
  const int half = lane >> 4;     // lane group
  const int m = lane & 15;        // A-matrix row within tile
  const int row = rowbase + m;

  // A fragments (16-bit A 16x32), contiguous K to match B:
  //  lanes 0-15 : elems 0..15 = K k0+0..15
  //  lanes 16-31: elems 0..15 = K k0+16..31
  v16bf afrag[4];
  const int kb = half ? 16 : 0;
#pragma unroll
  for (int kc = 0; kc < 4; ++kc) {
    const float* p = x + (size_t)row * D + kc * 32 + kb;
    float4 f0 = ((const float4*)p)[0];
    float4 f1 = ((const float4*)p)[1];
    float4 f2 = ((const float4*)(p + 8))[0];
    float4 f3 = ((const float4*)(p + 8))[1];
    v16bf a;
    a[0] = f2bf(f0.x);  a[1] = f2bf(f0.y);  a[2] = f2bf(f0.z);  a[3] = f2bf(f0.w);
    a[4] = f2bf(f1.x);  a[5] = f2bf(f1.y);  a[6] = f2bf(f1.z);  a[7] = f2bf(f1.w);
    a[8] = f2bf(f2.x);  a[9] = f2bf(f2.y);  a[10] = f2bf(f2.z); a[11] = f2bf(f2.w);
    a[12] = f2bf(f3.x); a[13] = f2bf(f3.y); a[14] = f2bf(f3.z); a[15] = f2bf(f3.w);
    afrag[kc] = a;
  }

  // B fragments: col n = lane&15; lanes 0-15 hold K k0+0..15 (elems 0..15),
  // lanes 16-31 hold K k0+16..31.  Contiguous in transposed LDS layout.
  const int n = lane & 15;
  const int kb2 = half ? 16 : 0;
#pragma unroll
  for (int nt = 0; nt < 8; ++nt) {
    const int n0 = nt * 16;
    v8f acc = {0.f, 0.f, 0.f, 0.f, 0.f, 0.f, 0.f, 0.f};
#pragma unroll
    for (int kc = 0; kc < 4; ++kc) {
      const v16bf b = *(const v16bf*)&Wt[(n0 + n) * D + kc * 32 + kb2];
      acc = __builtin_amdgcn_wmma_f32_16x16x32_bf16(
          false, afrag[kc], false, b, (short)0, acc, false, false);
    }
    // D layout (32-bit C/D 16x16): VGPR v -> row v (lanes 0-15) / v+8, col = lane&15
    float* hp = h + (size_t)(rowbase + half * 8) * D + n0 + n;
#pragma unroll
    for (int v = 0; v < 8; ++v) hp[(size_t)v * D] = acc[v];
  }
}

// ---------------------------------------------------------------------------
// alpha_l[n] = h[n] . att_l ; alpha_r[n] = h[n] . att_r
// ---------------------------------------------------------------------------
__global__ __launch_bounds__(256) void gcn_alpha_kernel(
    const float* __restrict__ h, const float* __restrict__ attl,
    const float* __restrict__ attr, float* __restrict__ al,
    float* __restrict__ ar) {
  int nid = blockIdx.x * 256 + threadIdx.x;
  if (nid >= N_NODES) return;
  const float4* hv = (const float4*)(h + (size_t)nid * D);
  const float4* lv = (const float4*)attl;
  const float4* rv = (const float4*)attr;
  float sl = 0.f, sr = 0.f;
#pragma unroll 8
  for (int i = 0; i < D / 4; ++i) {
    float4 a = hv[i], l = lv[i], r = rv[i];
    sl += a.x * l.x + a.y * l.y + a.z * l.z + a.w * l.w;
    sr += a.x * r.x + a.y * r.y + a.z * r.z + a.w * r.w;
  }
  al[nid] = sl;
  ar[nid] = sr;
}

// ---------------------------------------------------------------------------
// Pass A per edge: out-degree of source; segment max of relu(al[r]+ar[c]).
// relu >= 0 so integer-bit atomicMax against 0.0f init is order-correct.
// ---------------------------------------------------------------------------
__global__ __launch_bounds__(256) void gcn_edge_a_kernel(
    const int* __restrict__ src, const int* __restrict__ dst,
    const float* __restrict__ al, const float* __restrict__ ar,
    float* __restrict__ deg, float* __restrict__ amax) {
  int e = blockIdx.x * 256 + threadIdx.x;
  if (e >= N_EDGES) return;
  int r = src[e], c = dst[e];
  atomicAdd(&deg[r], 1.0f);
  float a = fmaxf(al[r] + ar[c], 0.0f);
  atomicMax((int*)&amax[c], __float_as_int(a));
}

// deg -> deg^{-1/2} in place
__global__ __launch_bounds__(256) void gcn_dis_kernel(float* __restrict__ deg) {
  int nid = blockIdx.x * 256 + threadIdx.x;
  if (nid >= N_NODES) return;
  float d = deg[nid];
  deg[nid] = d > 0.f ? rsqrtf(d) : 0.f;
}

// Pass B per edge: a_sum[c] += exp(a - a_max[c])
__global__ __launch_bounds__(256) void gcn_edge_b_kernel(
    const int* __restrict__ src, const int* __restrict__ dst,
    const float* __restrict__ al, const float* __restrict__ ar,
    const float* __restrict__ amax, float* __restrict__ asum) {
  int e = blockIdx.x * 256 + threadIdx.x;
  if (e >= N_EDGES) return;
  int r = src[e], c = dst[e];
  float a = fmaxf(al[r] + ar[c], 0.0f);
  atomicAdd(&asum[c], __expf(a - amax[c]));
}

// out = h (implements combine='add' as the scatter base)
__global__ __launch_bounds__(256) void gcn_copy_kernel(
    const float* __restrict__ h, float* __restrict__ out) {
  int i = blockIdx.x * 256 + threadIdx.x;
  if (i >= (N_NODES * D) / 4) return;
  ((float4*)out)[i] = ((const float4*)h)[i];
}

// Pass C: one wave per edge; lane handles 4 dims.
// out[c] += dis[r]*dis[c] * softmax_weight * h[r]
__global__ __launch_bounds__(256) void gcn_edge_c_kernel(
    const int* __restrict__ src, const int* __restrict__ dst,
    const float* __restrict__ dis, const float* __restrict__ al,
    const float* __restrict__ ar, const float* __restrict__ amax,
    const float* __restrict__ asum, const float* __restrict__ h,
    float* __restrict__ out) {
  int e = (blockIdx.x * 256 + threadIdx.x) >> 5;
  int lane = threadIdx.x & 31;
  if (e >= N_EDGES) return;
  int r = src[e], c = dst[e];
  float a = fmaxf(al[r] + ar[c], 0.0f);
  float coeff = dis[r] * dis[c] * __expf(a - amax[c]) / asum[c];
  const float4 hv = *(const float4*)(h + (size_t)r * D + lane * 4);
  float* op = out + (size_t)c * D + lane * 4;
  atomicAdd(op + 0, coeff * hv.x);
  atomicAdd(op + 1, coeff * hv.y);
  atomicAdd(op + 2, coeff * hv.z);
  atomicAdd(op + 3, coeff * hv.w);
}

extern "C" void kernel_launch(void* const* d_in, const int* in_sizes, int n_in,
                              void* d_out, int out_size, void* d_ws, size_t ws_size,
                              hipStream_t stream) {
  const float* x    = (const float*)d_in[0];
  const int*   ei   = (const int*)d_in[1];   // edge_index [2, E] (int32 on device)
  const float* W    = (const float*)d_in[2];
  const float* attl = (const float*)d_in[3];
  const float* attr = (const float*)d_in[4];
  float* out = (float*)d_out;

  const int* src = ei;              // row 0 = sources (j)
  const int* dst = ei + N_EDGES;    // row 1 = targets (i)

  // Workspace: h [N*D] | deg/dis [N] | al [N] | ar [N] | amax [N] | asum [N]
  float* h    = (float*)d_ws;
  float* deg  = h + (size_t)N_NODES * D;
  float* al   = deg + N_NODES;
  float* ar   = al + N_NODES;
  float* amax = ar + N_NODES;
  float* asum = amax + N_NODES;
  (void)ws_size; (void)in_sizes; (void)n_in; (void)out_size;

  // zero deg/al/ar/amax/asum (amax init 0.0f is valid: relu values >= 0)
  hipMemsetAsync(deg, 0, sizeof(float) * 5 * N_NODES, stream);

  gcn_gemm_kernel<<<(N_NODES / 16 + 7) / 8, 256, 0, stream>>>(x, W, h);
  gcn_alpha_kernel<<<(N_NODES + 255) / 256, 256, 0, stream>>>(h, attl, attr, al, ar);
  gcn_edge_a_kernel<<<(N_EDGES + 255) / 256, 256, 0, stream>>>(src, dst, al, ar, deg, amax);
  gcn_dis_kernel<<<(N_NODES + 255) / 256, 256, 0, stream>>>(deg);
  gcn_edge_b_kernel<<<(N_EDGES + 255) / 256, 256, 0, stream>>>(src, dst, al, ar, amax, asum);
  gcn_copy_kernel<<<((N_NODES * D / 4) + 255) / 256, 256, 0, stream>>>(h, out);
  gcn_edge_c_kernel<<<(N_EDGES * 32) / 256, 256, 0, stream>>>(
      src, dst, deg, al, ar, amax, asum, h, out);
}
